// BallAttention_10995116278111
// MI455X (gfx1250) — matrix operand order repaired and checked by reading, not compiled, not run on hardware
//
#include <hip/hip_runtime.h>
#include <hip/hip_bf16.h>

typedef __attribute__((ext_vector_type(16))) _Float16 v16h;
typedef __attribute__((ext_vector_type(8)))  _Float16 v8h;
typedef __attribute__((ext_vector_type(8)))  float    v8f;

// __builtin_amdgcn_cvt_pkrtz returns __fp16-based vectors (distinct clang type
// from _Float16 vectors, same layout). Use __fp16 typedefs on the store path.
typedef __attribute__((ext_vector_type(2)))  __fp16   v2y;
typedef __attribute__((ext_vector_type(4)))  __fp16   v4y;

#define BALL 128
#define DH   64

// ds_swizzle group-of-32 encodings: offset[15]=0, xor_mask[14:10], and_mask[4:0]=0x1f
#define SWZ_SWAPX1 0x041F
#define SWZ_SWAPX2 0x081F
#define SWZ_SWAPX4 0x101F
#define SWZ_SWAPX8 0x201F

#define SWZ_F32(x, imm) \
    __int_as_float(__builtin_amdgcn_ds_swizzle(__float_as_int(x), (imm)))

// One block = one ball. 256 threads = 8 wave32's. Wave w owns query rows
// [16w, 16w+16). Both GEMMs use v_wmma_f32_16x16x32_f16.
__global__ __launch_bounds__(256)
void ball_attn_wmma_kernel(const float* __restrict__ q,
                           const float* __restrict__ k,
                           const float* __restrict__ v,
                           float* __restrict__ out) {
    // sQK: Q rows [0..8191], K rows [8192..16383] (f16, row-major 128x64 each).
    // After the S phase it is reused as P storage: 8 waves * (16x128) halves.
    __shared__ _Float16 sQK[2 * BALL * DH];   // 32 KB
    __shared__ _Float16 sVt[DH * BALL];       // 16 KB, transposed: Vt[d][pos]

    const int tid  = threadIdx.x;
    const int lane = tid & 31;
    const int warp = tid >> 5;          // 0..7
    const int half = lane >> 4;         // 0 or 1 (lane group)
    const int l16  = lane & 15;

    const size_t ballBase = (size_t)blockIdx.x * (BALL * DH);
    const float* qg = q + ballBase;
    const float* kg = k + ballBase;
    const float* vg = v + ballBase;

    // ---- Stage Q,K: global f32 -> LDS f16 via packed cvt ----
#pragma unroll
    for (int i = 0; i < 8; ++i) {
        int idx = i * 1024 + tid * 4;            // 0..8191, 16B-aligned
        float4 fq = *(const float4*)(qg + idx);
        float4 fk = *(const float4*)(kg + idx);
        v2y q01 = __builtin_amdgcn_cvt_pkrtz(fq.x, fq.y);
        v2y q23 = __builtin_amdgcn_cvt_pkrtz(fq.z, fq.w);
        v2y k01 = __builtin_amdgcn_cvt_pkrtz(fk.x, fk.y);
        v2y k23 = __builtin_amdgcn_cvt_pkrtz(fk.z, fk.w);
        *(v4y*)(&sQK[idx])             = __builtin_shufflevector(q01, q23, 0, 1, 2, 3);
        *(v4y*)(&sQK[BALL * DH + idx]) = __builtin_shufflevector(k01, k23, 0, 1, 2, 3);
    }

    // ---- Stage V transposed: thread takes 2 consecutive positions x 4 d's,
    //      packs (pos, pos+1) into one half2 store per d (Vt rows contiguous
    //      in pos). 1024 tasks = 64 pos-pairs x 16 d-groups.
#pragma unroll
    for (int i = 0; i < 4; ++i) {
        int t   = i * 256 + tid;
        int pp  = t >> 4;              // pos pair 0..63
        int dg  = t & 15;              // d group of 4
        int pos = pp * 2;
        float4 f0 = *(const float4*)(vg + pos * DH       + dg * 4);
        float4 f1 = *(const float4*)(vg + (pos + 1) * DH + dg * 4);
        *(v2y*)(&sVt[(dg * 4 + 0) * BALL + pos]) = __builtin_amdgcn_cvt_pkrtz(f0.x, f1.x);
        *(v2y*)(&sVt[(dg * 4 + 1) * BALL + pos]) = __builtin_amdgcn_cvt_pkrtz(f0.y, f1.y);
        *(v2y*)(&sVt[(dg * 4 + 2) * BALL + pos]) = __builtin_amdgcn_cvt_pkrtz(f0.z, f1.z);
        *(v2y*)(&sVt[(dg * 4 + 3) * BALL + pos]) = __builtin_amdgcn_cvt_pkrtz(f0.w, f1.w);
    }
    __syncthreads();

    const _Float16* sQ = sQK;
    const _Float16* sK = sQK + BALL * DH;

    // ---- Load Q A-fragments: 16x32 chunks c=0,1 (d0=0,32) ----
    // A fragment, contiguous k order (not the ISA table order): lanes 0-15 row=l16 halves0..15=K d0..d0+15;
    // lanes 16-31 same row, K d0+16..d0+31 (matches contiguous B).
    const int qrow = warp * 16 + l16;
    v16h aQ[2];
#pragma unroll
    for (int c = 0; c < 2; ++c) {
        int d0 = c * 32;
        aQ[c] = *(const v16h*)(&sQ[qrow * DH + d0 + half * 16]);
    }

    // ---- S = (Q K^T) * scale : 8 column tiles, K-dim = 64 (2 WMMAs each) ----
    v8f accS[8];
#pragma unroll
    for (int j = 0; j < 8; ++j) {
        v8f acc = {};
        int col = j * 16 + l16;                    // key index for B fragment
#pragma unroll
        for (int c = 0; c < 2; ++c) {
            int d0 = c * 32;
            // B fragment, same contiguous k order as A: 16 contiguous halves at k = d0 + half*16
            v16h bK = *(const v16h*)(&sK[col * DH + d0 + half * 16]);
            acc = __builtin_amdgcn_wmma_f32_16x16x32_f16(
                false, aQ[c], false, bK, (short)0, acc, false, false);
        }
        accS[j] = acc;
    }

    // ---- Softmax over 128 keys, rows r (lanes 0-15) / r+8 (lanes 16-31).
    //      Row values live across one 16-lane half -> 4 ds_swizzle butterflies.
    const float scale = 0.125f;   // 1/sqrt(64)
#pragma unroll
    for (int r = 0; r < 8; ++r) {
        float m = -1e30f;
#pragma unroll
        for (int j = 0; j < 8; ++j) {
            accS[j][r] *= scale;
            m = fmaxf(m, accS[j][r]);
        }
        m = fmaxf(m, SWZ_F32(m, SWZ_SWAPX1));
        m = fmaxf(m, SWZ_F32(m, SWZ_SWAPX2));
        m = fmaxf(m, SWZ_F32(m, SWZ_SWAPX4));
        m = fmaxf(m, SWZ_F32(m, SWZ_SWAPX8));
        float sum = 0.f;
#pragma unroll
        for (int j = 0; j < 8; ++j) {
            accS[j][r] = __expf(accS[j][r] - m);
            sum += accS[j][r];
        }
        sum += SWZ_F32(sum, SWZ_SWAPX1);
        sum += SWZ_F32(sum, SWZ_SWAPX2);
        sum += SWZ_F32(sum, SWZ_SWAPX4);
        sum += SWZ_F32(sum, SWZ_SWAPX8);
        float inv = 1.0f / sum;
#pragma unroll
        for (int j = 0; j < 8; ++j)
            accS[j][r] *= inv;
    }

    // ---- All waves done reading Q/K; reuse sQK as per-wave P (16x128 f16) ----
    __syncthreads();
    _Float16* sP = sQK + warp * (16 * BALL);
#pragma unroll
    for (int j = 0; j < 8; ++j) {
#pragma unroll
        for (int r = 0; r < 8; ++r) {
            int rowl = r + 8 * half;               // D-layout row within tile
            sP[rowl * BALL + j * 16 + l16] = (_Float16)accS[j][r];
        }
    }
    // Same-wave LDS RAW is in-order; each wave reads only its own P region.

    // ---- Load P A-fragments: 4 chunks over k=0..127 ----
    v16h aP[4];
#pragma unroll
    for (int t = 0; t < 4; ++t) {
        int k0 = t * 32;
        aP[t] = *(const v16h*)(&sP[l16 * BALL + k0 + half * 16]);
    }

    // ---- O = P V : 4 output tiles over Dh, K-dim = 128 (4 WMMAs each) ----
    float* og = out + ballBase;
#pragma unroll
    for (int ct = 0; ct < 4; ++ct) {
        v8f acc = {};
        int dcol = ct * 16 + l16;                  // output feature column
#pragma unroll
        for (int t = 0; t < 4; ++t) {
            int k0 = t * 32;
            // B fragment from Vt: 16 contiguous halves at k = k0 + half*16
            v16h bV = *(const v16h*)(&sVt[dcol * BALL + k0 + half * 16]);
            acc = __builtin_amdgcn_wmma_f32_16x16x32_f16(
                false, aP[t], false, bV, (short)0, acc, false, false);
        }
#pragma unroll
        for (int r = 0; r < 8; ++r) {
            int row = warp * 16 + r + 8 * half;
            og[row * DH + dcol] = acc[r];
        }
    }
}

extern "C" void kernel_launch(void* const* d_in, const int* in_sizes, int n_in,
                              void* d_out, int out_size, void* d_ws, size_t ws_size,
                              hipStream_t stream) {
    (void)n_in; (void)out_size; (void)d_ws; (void)ws_size;
    const float* q = (const float*)d_in[0];
    const float* k = (const float*)d_in[1];
    const float* v = (const float*)d_in[2];
    float* out = (float*)d_out;
    int balls = in_sizes[0] / (BALL * DH);   // B*H*(n/BALL) = 4096
    ball_attn_wmma_kernel<<<balls, 256, 0, stream>>>(q, k, v, out);
}
